// MultiLayerLSTM_56289841381919
// MI455X (gfx1250) — matrix operand order repaired and checked by reading, not compiled, not run on hardware
//
#include <hip/hip_runtime.h>
#include <hip/hip_bf16.h>

typedef __bf16 bf16_t;
typedef __attribute__((ext_vector_type(16))) __bf16 v16bf;
typedef __attribute__((ext_vector_type(8)))  float  v8f;

#define WMMA_BF16(a, b, c) \
  __builtin_amdgcn_wmma_f32_16x16x32_bf16(false, (a), false, (b), (short)0, (c), false, false)

__device__ __forceinline__ bf16_t f2bf(float f) {
  unsigned int u = __float_as_uint(f);
  unsigned int r = (u + 0x7FFFu + ((u >> 16) & 1u)) >> 16;   // round-to-nearest-even
  unsigned short s = (unsigned short)r;
  return __builtin_bit_cast(bf16_t, s);
}

__device__ __forceinline__ float sigf(float x) { return 1.0f / (1.0f + __expf(-x)); }

union AFrag { v16bf v; unsigned int u[8]; };

// A fragment (16x32 bf16): lane%16 = row M, lane/16 selects +8 within each K-16 group.
// VGPR v<4: K = 2v,2v+1 (+8*half); v>=4: K = 16+2(v-4),... (+8*half)
__device__ __forceinline__ v16bf load_a_frag(const bf16_t* __restrict__ act, long rowStride,
                                             int m0, int lane, int kbase) {
  AFrag a;
  const bf16_t* row = act + (long)(m0 + (lane & 15)) * rowStride + kbase + ((lane >> 4) << 3);
#pragma unroll
  for (int v = 0; v < 8; ++v) {
    int k = (v < 4) ? (v * 2) : (16 + (v - 4) * 2);
    a.u[v] = *(const unsigned int*)(row + k);
  }
  return a.v;
}

// B fragment: pre-swizzled, one contiguous 32B load per lane.
__device__ __forceinline__ v16bf load_b_frag(const bf16_t* __restrict__ P, int nK,
                                             int nt, int kt, int lane) {
  return *(const v16bf*)(P + (((long)(nt * nK + kt) * 32 + lane) << 4));
}

// ---------- prep kernels ----------
__global__ void cvt_f32_bf16(const float* __restrict__ s, bf16_t* __restrict__ d, long n) {
  long i = (long)blockIdx.x * blockDim.x + threadIdx.x;
  if (i < n) d[i] = f2bf(s[i]);
}
__global__ void zero_u16(unsigned short* p, long n) {
  long i = (long)blockIdx.x * blockDim.x + threadIdx.x;
  if (i < n) p[i] = 0;
}
__global__ void zero_f32(float* p, long n) {
  long i = (long)blockIdx.x * blockDim.x + threadIdx.x;
  if (i < n) p[i] = 0.0f;
}

// Swizzle W[K][N] (row-major f32) into WMMA B-fragment order (bf16):
// B 32x16 bf16: lane%16 = col n; v<4: K = (lane/16)*8 + 2v + {0,1}; v>=4: K = 16 + (lane/16)*8 + 2(v-4) + {0,1}
__global__ void swizzle_b_kernel(const float* __restrict__ W, bf16_t* __restrict__ P,
                                 int N, int K) {
  int nK = K / 32;
  long gid = (long)blockIdx.x * blockDim.x + threadIdx.x;
  long total = (long)(N / 16) * nK * 256;  // 32 lanes * 8 pairs per tile
  if (gid >= total) return;
  int v    = (int)(gid & 7);
  int lane = (int)((gid >> 3) & 31);
  long tile = gid >> 8;                    // tile = nt*nK + kt
  int kt = (int)(tile % nK);
  int n  = (int)(tile / nK) * 16 + (lane & 15);
  int k  = kt * 32 + ((lane >> 4) << 3) + ((v < 4) ? (v * 2) : (16 + (v - 4) * 2));
  long dst = (tile * 32 + lane) * 16 + v * 2;
  P[dst]     = f2bf(W[(long)k * N + n]);
  P[dst + 1] = f2bf(W[(long)(k + 1) * N + n]);
}

// ---------- LSTM cell: one (t, layer) ----------
// grid = H/16 = 64 blocks, block = 32 threads (one wave per column-slice).
// Each wave: 2 M-tiles x 4 gates, K loop over x-part then h-part.
__global__ __launch_bounds__(32) void lstm_cell_kernel(
    const bf16_t* __restrict__ Ax, long sx, int nKx, const bf16_t* __restrict__ Wxp,
    const bf16_t* __restrict__ Ah, const bf16_t* __restrict__ Whp,
    const float* __restrict__ bias, float* __restrict__ cbuf, bf16_t* __restrict__ hout)
{
  const int H = 1024;
  const int nKh = H / 32;         // 32
  const int nSlice = H / 16;      // 64
  int lane  = threadIdx.x;
  int slice = blockIdx.x;

  v8f acc[2][4];
#pragma unroll
  for (int mt = 0; mt < 2; ++mt)
#pragma unroll
    for (int g = 0; g < 4; ++g) acc[mt][g] = (v8f){0,0,0,0,0,0,0,0};

  for (int kt = 0; kt < nKx; ++kt) {
    v16bf a0 = load_a_frag(Ax, sx, 0,  lane, kt * 32);
    v16bf a1 = load_a_frag(Ax, sx, 16, lane, kt * 32);
#pragma unroll
    for (int g = 0; g < 4; ++g) {
      v16bf bm = load_b_frag(Wxp, nKx, g * nSlice + slice, kt, lane);
      acc[0][g] = WMMA_BF16(a0, bm, acc[0][g]);
      acc[1][g] = WMMA_BF16(a1, bm, acc[1][g]);
    }
  }
  for (int kt = 0; kt < nKh; ++kt) {
    v16bf a0 = load_a_frag(Ah, H, 0,  lane, kt * 32);
    v16bf a1 = load_a_frag(Ah, H, 16, lane, kt * 32);
#pragma unroll
    for (int g = 0; g < 4; ++g) {
      v16bf bm = load_b_frag(Whp, nKh, g * nSlice + slice, kt, lane);
      acc[0][g] = WMMA_BF16(a0, bm, acc[0][g]);
      acc[1][g] = WMMA_BF16(a1, bm, acc[1][g]);
    }
  }

  // C layout: n = lane%16, m = v + 8*(lane/16) within the M-tile
  int col  = slice * 16 + (lane & 15);
  int half = lane >> 4;
  float bi = bias[col], bf = bias[H + col], bg = bias[2 * H + col], bo = bias[3 * H + col];
#pragma unroll
  for (int mt = 0; mt < 2; ++mt) {
#pragma unroll
    for (int v = 0; v < 8; ++v) {
      int m = mt * 16 + v + half * 8;
      long idx = (long)m * H + col;
      float iv = sigf(acc[mt][0][v] + bi);
      float fv = sigf(acc[mt][1][v] + bf);
      float gv = tanhf(acc[mt][2][v] + bg);
      float ov = sigf(acc[mt][3][v] + bo);
      float cn = fv * cbuf[idx] + iv * gv;
      cbuf[idx] = cn;
      hout[idx] = f2bf(ov * tanhf(cn));
    }
  }
}

// ---------- output projection: h_final[32x1024] @ Wout[1024x512] + bout ----------
__global__ __launch_bounds__(32) void proj_kernel(
    const bf16_t* __restrict__ hfin, const bf16_t* __restrict__ Wp,
    const float* __restrict__ bout, float* __restrict__ out)
{
  const int H = 1024, DOUT = 512;
  const int nK = H / 32;
  int lane = threadIdx.x;
  int nt = blockIdx.x;  // 0..31
  v8f acc0 = (v8f){0,0,0,0,0,0,0,0};
  v8f acc1 = acc0;
  for (int kt = 0; kt < nK; ++kt) {
    v16bf a0 = load_a_frag(hfin, H, 0,  lane, kt * 32);
    v16bf a1 = load_a_frag(hfin, H, 16, lane, kt * 32);
    v16bf bm = load_b_frag(Wp, nK, nt, kt, lane);
    acc0 = WMMA_BF16(a0, bm, acc0);
    acc1 = WMMA_BF16(a1, bm, acc1);
  }
  int n = nt * 16 + (lane & 15);
  int half = lane >> 4;
  float bb = bout[n];
#pragma unroll
  for (int v = 0; v < 8; ++v) {
    out[(long)(v + half * 8) * DOUT + n]      = acc0[v] + bb;
    out[(long)(16 + v + half * 8) * DOUT + n] = acc1[v] + bb;
  }
}

extern "C" void kernel_launch(void* const* d_in, const int* in_sizes, int n_in,
                              void* d_out, int out_size, void* d_ws, size_t ws_size,
                              hipStream_t stream) {
  (void)in_sizes; (void)n_in; (void)out_size; (void)ws_size;
  const int Bb = 32, T = 512, DIN = 512, H = 1024, L = 4, DOUT = 512, G = 4 * H;

  const float* x    = (const float*)d_in[0];
  const float* Wx0  = (const float*)d_in[1];
  const float* Wh0  = (const float*)d_in[2];
  const float* b0   = (const float*)d_in[3];
  const float* Wx   = (const float*)d_in[4];
  const float* Wh   = (const float*)d_in[5];
  const float* b    = (const float*)d_in[6];
  const float* Wout = (const float*)d_in[7];
  const float* bout = (const float*)d_in[8];
  float* out = (float*)d_out;

  char* ws = (char*)d_ws;
  size_t off = 0;
  auto take = [&](size_t bytes) {
    char* p = ws + off;
    off = (off + bytes + 255) & ~(size_t)255;
    return p;
  };
  bf16_t* Xbf   = (bf16_t*)take((size_t)Bb * T * DIN * 2);     // 16 MB
  bf16_t* Wx0p  = (bf16_t*)take((size_t)DIN * G * 2);          //  4 MB
  bf16_t* Wh0p  = (bf16_t*)take((size_t)H * G * 2);            //  8 MB
  bf16_t* Wxp   = (bf16_t*)take((size_t)3 * H * G * 2);        // 24 MB
  bf16_t* Whp   = (bf16_t*)take((size_t)3 * H * G * 2);        // 24 MB
  bf16_t* Woutp = (bf16_t*)take((size_t)H * DOUT * 2);         //  1 MB
  bf16_t* hbuf  = (bf16_t*)take((size_t)2 * L * Bb * H * 2);   // double-buffered h (bf16)
  float*  cbuf  = (float*) take((size_t)L * Bb * H * 4);       // cell state (f32)

  long nX = (long)Bb * T * DIN;
  cvt_f32_bf16<<<dim3((unsigned)((nX + 255) / 256)), dim3(256), 0, stream>>>(x, Xbf, nX);

  auto swz = [&](const float* W, bf16_t* P, int N, int K) {
    long total = (long)(N / 16) * (K / 32) * 256;
    swizzle_b_kernel<<<dim3((unsigned)((total + 255) / 256)), dim3(256), 0, stream>>>(W, P, N, K);
  };
  swz(Wx0, Wx0p, G, DIN);
  swz(Wh0, Wh0p, G, H);
  for (int i = 0; i < 3; ++i) {
    swz(Wx + (size_t)i * H * G, Wxp + (size_t)i * H * G, G, H);
    swz(Wh + (size_t)i * H * G, Whp + (size_t)i * H * G, G, H);
  }
  swz(Wout, Woutp, DOUT, H);

  long nH = (long)2 * L * Bb * H;
  zero_u16<<<dim3((unsigned)((nH + 255) / 256)), dim3(256), 0, stream>>>((unsigned short*)hbuf, nH);
  long nC = (long)L * Bb * H;
  zero_f32<<<dim3((unsigned)((nC + 255) / 256)), dim3(256), 0, stream>>>(cbuf, nC);

  const size_t ls = (size_t)Bb * H;  // elems per layer in h/c buffers
  for (int t = 0; t < T; ++t) {
    int rb = t & 1, wb = rb ^ 1;     // read t-1 h from rb, write t h to wb
    for (int l = 0; l < L; ++l) {
      const bf16_t* Ax; long sx; int nKx; const bf16_t* WxpL; const float* bias;
      if (l == 0) {
        Ax = Xbf + (size_t)t * DIN; sx = (long)T * DIN; nKx = DIN / 32;
        WxpL = Wx0p; bias = b0;
      } else {
        Ax = hbuf + ((size_t)wb * L + (l - 1)) * ls; sx = H; nKx = H / 32;
        WxpL = Wxp + (size_t)(l - 1) * H * G; bias = b + (size_t)(l - 1) * G;
      }
      const bf16_t* WhpL = (l == 0) ? Wh0p : (Whp + (size_t)(l - 1) * H * G);
      const bf16_t* Ah   = hbuf + ((size_t)rb * L + l) * ls;
      bf16_t*       hout = hbuf + ((size_t)wb * L + l) * ls;
      float*        cL   = cbuf + (size_t)l * ls;
      lstm_cell_kernel<<<dim3(H / 16), dim3(32), 0, stream>>>(
          Ax, sx, nKx, WxpL, Ah, WhpL, bias, cL, hout);
    }
  }

  // t = T-1 = 511 wrote into buffer wb = 0; project last layer's h.
  const bf16_t* hfin = hbuf + ((size_t)0 * L + (L - 1)) * ls;
  proj_kernel<<<dim3(DOUT / 16), dim3(32), 0, stream>>>(hfin, Woutp, bout, out);
}
